// Attention_3831110828630
// MI455X (gfx1250) — matrix operand order repaired and checked by reading, not compiled, not run on hardware
//
#include <hip/hip_runtime.h>
#include <hip/hip_bf16.h>

// ---------------------------------------------------------------------------
// MI455X (gfx1250) attention: bf16 WMMA for all GEMM stages, f32 softmax.
//   B=4, N=2048, DIM=768, H=12, hd=64, SCALE=1/8
// Round 3: software-pipelined GEMMs with FULLY UNROLLED k-loops so the
// next-step fragment loads become pure SSA (no loop-carried register copies
// — round 2 showed the allocator cannot coalesce them), WMMAs consume
// older loads while newer ones are in flight. Attention KV loop unrolled 2x
// to halve back-edge copy overhead; global_prefetch pulls upcoming tiles.
// ---------------------------------------------------------------------------

typedef __attribute__((ext_vector_type(16))) __bf16 v16bf;
typedef __attribute__((ext_vector_type(8)))  __bf16 v8bf;
typedef __attribute__((ext_vector_type(8)))  float  v8f;

#define BATCH 4
#define SEQ   2048
#define DIM   768
#define NH    12
#define HD    64

__device__ __forceinline__ v8f wmma_bf16(v16bf a, v16bf b, v8f c) {
  // D = A(16x32) * B(32x16) + C, f32 accumulate
  return __builtin_amdgcn_wmma_f32_16x16x32_bf16(
      /*neg_a=*/false, a, /*neg_b=*/false, b,
      /*c_mod=*/(short)0, c, /*reuse_a=*/false, /*reuse_b=*/false);
}

// A fragment: 16x32 bf16, source row-major [16][ld] at `base`.
// Lane l (l<16) holds row m=l, K = {hf*8 + e} and {16 + hf*8 + e}.
__device__ __forceinline__ v16bf load_a_frag(const __bf16* base, int ld, int lane) {
  const int hf = (lane >> 4) & 1;
  const int m  = lane & 15;
  const __bf16* p = base + (size_t)m * ld + hf * 8;
  v8bf lo = *(const v8bf*)(p);        // K = hf*8 + 0..7
  v8bf hi = *(const v8bf*)(p + 16);   // K = 16 + hf*8 + 0..7
  return __builtin_shufflevector(lo, hi, 0,1,2,3,4,5,6,7,8,9,10,11,12,13,14,15);
}

// B fragment: 32x16 bf16, B[kk][n] = Mat[n][kk] with Mat row-major [16][ld]
// at `base`. Lane holds column n=lane&15, K = {hf*8 + e} and {16 + hf*8 + e}.
__device__ __forceinline__ v16bf load_b_frag(const __bf16* base, int ld, int lane) {
  const int hf = (lane >> 4) & 1;
  const int n  = lane & 15;
  const __bf16* p = base + (size_t)n * ld + hf * 8;
  v8bf lo = *(const v8bf*)(p);
  v8bf hi = *(const v8bf*)(p + 16);
  return __builtin_shufflevector(lo, hi, 0,1,2,3,4,5,6,7,8,9,10,11,12,13,14,15);
}

// ---------------------------------------------------------------------------
__global__ void cvt_f32_bf16_kernel(const float* __restrict__ src,
                                    __bf16* __restrict__ dst, int n) {
  int i = blockIdx.x * blockDim.x + threadIdx.x;
  int stride = gridDim.x * blockDim.x;
  for (; i < n; i += stride) dst[i] = (__bf16)src[i];
}

// ---------------------------------------------------------------------------
// QKV GEMM: qkv[m,n] = sum_k xb[m,k]*wq[n,k] (+bias). Wave: 32 rows x 64 cols
// (8 WMMA / k-step, B fragments reused across both row tiles). k-loop fully
// unrolled with next-step loads issued ahead of current-step WMMAs.
// Epilogue scatters into q_scaled [BH][N][hd], k [BH][N][hd], vT [BH][hd][N].
__global__ void __launch_bounds__(256)
qkv_gemm_kernel(const __bf16* __restrict__ xb, const __bf16* __restrict__ wq,
                const float* __restrict__ q_bias, const float* __restrict__ v_bias,
                __bf16* __restrict__ qs, __bf16* __restrict__ kbuf,
                __bf16* __restrict__ vT) {
  const int lane = threadIdx.x & 31;
  const int wave = threadIdx.x >> 5;
  const int hf   = lane >> 4;
  const int ln   = lane & 15;
  const int wm   = wave & 3;                       // 4 waves along M
  const int wn   = wave >> 2;                      // 2 waves along N
  const int m0 = blockIdx.y * 128 + wm * 32;
  const int n0 = blockIdx.x * 128 + wn * 64;

  const __bf16* Abase = xb + (size_t)m0 * DIM;
  const __bf16* Bbase = wq + (size_t)n0 * DIM;

  v8f acc[2][4];
#pragma unroll
  for (int mi = 0; mi < 2; ++mi)
#pragma unroll
    for (int t = 0; t < 4; ++t)
#pragma unroll
      for (int j = 0; j < 8; ++j) acc[mi][t][j] = 0.0f;

  // prologue: fragments for k0 = 0
  v16bf a[2], b[4];
#pragma unroll
  for (int mi = 0; mi < 2; ++mi)
    a[mi] = load_a_frag(Abase + (size_t)mi * 16 * DIM, DIM, lane);
#pragma unroll
  for (int t = 0; t < 4; ++t)
    b[t] = load_b_frag(Bbase + (size_t)t * 16 * DIM, DIM, lane);

#pragma unroll
  for (int k0 = 0; k0 < DIM; k0 += 32) {
    // issue next k-step fragment loads ahead of this step's WMMAs
    // (condition is compile-time after full unroll; folds away on last step)
    v16bf na[2] = {a[0], a[1]};
    v16bf nb[4] = {b[0], b[1], b[2], b[3]};
    if (k0 + 32 < DIM) {
      const int kn = k0 + 32;
#pragma unroll
      for (int mi = 0; mi < 2; ++mi)
        na[mi] = load_a_frag(Abase + (size_t)mi * 16 * DIM + kn, DIM, lane);
#pragma unroll
      for (int t = 0; t < 4; ++t)
        nb[t] = load_b_frag(Bbase + (size_t)t * 16 * DIM + kn, DIM, lane);
    }

#pragma unroll
    for (int t = 0; t < 4; ++t) {
      acc[0][t] = wmma_bf16(a[0], b[t], acc[0][t]);
      acc[1][t] = wmma_bf16(a[1], b[t], acc[1][t]);
    }
#pragma unroll
    for (int mi = 0; mi < 2; ++mi) a[mi] = na[mi];
#pragma unroll
    for (int t = 0; t < 4; ++t) b[t] = nb[t];
  }

#pragma unroll
  for (int t = 0; t < 4; ++t) {
    const int n     = n0 + t * 16;    // tile col base; lane adds ln
    const int which = n / DIM;        // 0=q, 1=k, 2=v (uniform per tile)
    const int c0    = n % DIM;
    const int h     = c0 / HD;        // head (uniform per tile)
    const int d     = (c0 % HD) + ln; // hd index for this lane
    const int cfull = c0 + ln;        // bias index
#pragma unroll
    for (int mi = 0; mi < 2; ++mi)
#pragma unroll
      for (int j = 0; j < 8; ++j) {
        const int row  = m0 + mi * 16 + j + hf * 8;  // D layout: M=j+half*8
        const int bidx = row >> 11;                  // /SEQ
        const int tok  = row & (SEQ - 1);
        const int bh   = bidx * NH + h;
        float val = acc[mi][t][j];
        if (which == 0) {
          val = (val + q_bias[cfull]) * 0.125f;      // fold SCALE = hd^-0.5
          qs[(size_t)bh * SEQ * HD + (size_t)tok * HD + d] = (__bf16)val;
        } else if (which == 1) {
          kbuf[(size_t)bh * SEQ * HD + (size_t)tok * HD + d] = (__bf16)val;
        } else {
          val = val + v_bias[cfull];
          vT[(size_t)bh * HD * SEQ + (size_t)d * SEQ + tok] = (__bf16)val;
        }
      }
  }
}

// ---------------------------------------------------------------------------
// Flash attention: wave owns 16 query rows; 32 keys/iter:
//   S = Q(16x64)·K^T -> 4 wmma;  P·V -> 4 wmma (P staged via LDS transpose).
// K fragments for iter i+1 issued right after the S-WMMAs consume iter i's
// (softmax VALU overlaps the loads); V fragments double-buffered across the
// s_wait_dscnt point. Loop unrolled 2x so intra-body rotations rename away.
__global__ void __launch_bounds__(256)
attn_kernel(const __bf16* __restrict__ qs, const __bf16* __restrict__ kbuf,
            const __bf16* __restrict__ vT, const int* __restrict__ mask,
            __bf16* __restrict__ ao) {
  __shared__ __bf16 Pb[8][16][32];   // per-wave 16x32 P staging (8 KB)

  const int lane = threadIdx.x & 31;
  const int wave = threadIdx.x >> 5;
  const int hf   = lane >> 4;
  const int ln   = lane & 15;
  const int bh = blockIdx.x >> 4;    // (b*NH + h)
  const int qt = blockIdx.x & 15;
  const int b  = bh / NH;
  const int h  = bh % NH;
  const int q0 = qt * 128 + wave * 16;

  const __bf16* qp = qs   + (size_t)bh * SEQ * HD + (size_t)q0 * HD;
  const __bf16* kp = kbuf + (size_t)bh * SEQ * HD;
  const __bf16* vp = vT   + (size_t)bh * HD * SEQ;
  const int*    mp = mask + (size_t)b * SEQ;

  // Q fragments held in registers for the whole KV sweep (hd = 2 x 32)
  const v16bf aq0 = load_a_frag(qp + 0,  HD, lane);
  const v16bf aq1 = load_a_frag(qp + 32, HD, lane);

  v8f o[4];
  float mrow[8], lrow[8];
#pragma unroll
  for (int j = 0; j < 8; ++j) { mrow[j] = -3.0e38f; lrow[j] = 0.0f; }
#pragma unroll
  for (int t = 0; t < 4; ++t)
#pragma unroll
    for (int j = 0; j < 8; ++j) o[t][j] = 0.0f;

  // prologue: K / V fragments for the first 32-key block
  v16bf kf[4], vf[4];
  kf[0] = load_b_frag(kp + (size_t) 0 * HD +  0, HD, lane);
  kf[1] = load_b_frag(kp + (size_t) 0 * HD + 32, HD, lane);
  kf[2] = load_b_frag(kp + (size_t)16 * HD +  0, HD, lane);
  kf[3] = load_b_frag(kp + (size_t)16 * HD + 32, HD, lane);
#pragma unroll
  for (int t = 0; t < 4; ++t)
    vf[t] = load_b_frag(vp + (size_t)(t * 16) * SEQ, SEQ, lane);

#pragma unroll 2
  for (int kb0 = 0; kb0 < SEQ; kb0 += 32) {
    v8f z;
#pragma unroll
    for (int j = 0; j < 8; ++j) z[j] = 0.0f;

    // S tiles: keys [kb0, kb0+16) and [kb0+16, kb0+32), contraction over hd
    v8f s0 = wmma_bf16(aq0, kf[0], z);
    s0     = wmma_bf16(aq1, kf[1], s0);
    v8f s1 = wmma_bf16(aq0, kf[2], z);
    s1     = wmma_bf16(aq1, kf[3], s1);

    // pipeline: issue next block's K fragments now (softmax hides latency)
    const int kn = (kb0 + 32 < SEQ) ? (kb0 + 32) : kb0;
    kf[0] = load_b_frag(kp + (size_t)(kn +  0) * HD +  0, HD, lane);
    kf[1] = load_b_frag(kp + (size_t)(kn +  0) * HD + 32, HD, lane);
    kf[2] = load_b_frag(kp + (size_t)(kn + 16) * HD +  0, HD, lane);
    kf[3] = load_b_frag(kp + (size_t)(kn + 16) * HD + 32, HD, lane);
    v16bf vfn[4];
#pragma unroll
    for (int t = 0; t < 4; ++t)
      vfn[t] = load_b_frag(vp + (size_t)(t * 16) * SEQ + kn, SEQ, lane);
    // prefetch two blocks ahead toward the WGP (CDNA5 global_prefetch_b8)
    const int kpf = (kb0 + 64) & (SEQ - 1);
    __builtin_prefetch(kp + (size_t)(kpf + (lane & 1) * 16) * HD, 0, 0);
    __builtin_prefetch(vp + (size_t)(lane & 3) * 16 * SEQ + kpf, 0, 0);

    // mask: D-layout column = lane&15
    const float madd0 = mp[kb0 + ln]      ? 0.0f : -1.0e30f;
    const float madd1 = mp[kb0 + 16 + ln] ? 0.0f : -1.0e30f;

#pragma unroll
    for (int j = 0; j < 8; ++j) {
      float v0 = s0[j] + madd0;
      float v1 = s1[j] + madd1;
      // row max across the 16 lanes of this half (row lives in one half)
      float r = fmaxf(v0, v1);
      r = fmaxf(r, __shfl_xor(r, 1, 32));
      r = fmaxf(r, __shfl_xor(r, 2, 32));
      r = fmaxf(r, __shfl_xor(r, 4, 32));
      r = fmaxf(r, __shfl_xor(r, 8, 32));
      const float mnew = fmaxf(mrow[j], r);
      const float corr = __expf(mrow[j] - mnew);
      const float p0 = __expf(v0 - mnew);
      const float p1 = __expf(v1 - mnew);
      float rs = p0 + p1;
      rs += __shfl_xor(rs, 1, 32);
      rs += __shfl_xor(rs, 2, 32);
      rs += __shfl_xor(rs, 4, 32);
      rs += __shfl_xor(rs, 8, 32);
      lrow[j] = lrow[j] * corr + rs;
      mrow[j] = mnew;
#pragma unroll
      for (int t = 0; t < 4; ++t) o[t][j] *= corr;
      // stage P (D layout -> row-major 16x32) for the A-fragment reload
      Pb[wave][j + hf * 8][ln]      = (__bf16)p0;
      Pb[wave][j + hf * 8][16 + ln] = (__bf16)p1;
    }

    // wave-local LDS visibility: CDNA5 split DS counter
    asm volatile("s_wait_dscnt 0" ::: "memory");

    // reload P as an A fragment (row m = lane&15)
    v16bf ap;
    {
      const __bf16* pp = &Pb[wave][ln][hf * 8];
#pragma unroll
      for (int e = 0; e < 8; ++e) { ap[e] = pp[e]; ap[e + 8] = pp[16 + e]; }
    }

    // O += P(16x32) · V(32x64), V streamed transposed [hd][N]
#pragma unroll
    for (int t = 0; t < 4; ++t) o[t] = wmma_bf16(ap, vf[t], o[t]);
#pragma unroll
    for (int t = 0; t < 4; ++t) vf[t] = vfn[t];
  }

  // normalize and write per-head output [B,N,DIM] (bf16 for proj GEMM)
#pragma unroll
  for (int j = 0; j < 8; ++j) {
    const float inv = 1.0f / fmaxf(lrow[j], 1e-20f);
#pragma unroll
    for (int t = 0; t < 4; ++t) o[t][j] *= inv;
  }
#pragma unroll
  for (int t = 0; t < 4; ++t)
#pragma unroll
    for (int j = 0; j < 8; ++j) {
      const int tok = q0 + j + hf * 8;
      const int col = h * HD + t * 16 + ln;
      ao[(size_t)(b * SEQ + tok) * DIM + col] = (__bf16)o[t][j];
    }
}

// ---------------------------------------------------------------------------
// Projection: out[m,n] = sum_k ao[m,k]*wp[n,k] + proj_b[n], f32 output.
// Same fully-unrolled pipelined 32x64 wave tile as the QKV GEMM.
__global__ void __launch_bounds__(256)
proj_gemm_kernel(const __bf16* __restrict__ ao, const __bf16* __restrict__ wp,
                 const float* __restrict__ proj_b, float* __restrict__ out) {
  const int lane = threadIdx.x & 31;
  const int wave = threadIdx.x >> 5;
  const int hf   = lane >> 4;
  const int ln   = lane & 15;
  const int wm   = wave & 3;
  const int wn   = wave >> 2;
  const int m0 = blockIdx.y * 128 + wm * 32;
  const int n0 = blockIdx.x * 128 + wn * 64;

  const __bf16* Abase = ao + (size_t)m0 * DIM;
  const __bf16* Bbase = wp + (size_t)n0 * DIM;

  v8f acc[2][4];
#pragma unroll
  for (int mi = 0; mi < 2; ++mi)
#pragma unroll
    for (int t = 0; t < 4; ++t)
#pragma unroll
      for (int j = 0; j < 8; ++j) acc[mi][t][j] = 0.0f;

  v16bf a[2], b[4];
#pragma unroll
  for (int mi = 0; mi < 2; ++mi)
    a[mi] = load_a_frag(Abase + (size_t)mi * 16 * DIM, DIM, lane);
#pragma unroll
  for (int t = 0; t < 4; ++t)
    b[t] = load_b_frag(Bbase + (size_t)t * 16 * DIM, DIM, lane);

#pragma unroll
  for (int k0 = 0; k0 < DIM; k0 += 32) {
    v16bf na[2] = {a[0], a[1]};
    v16bf nb[4] = {b[0], b[1], b[2], b[3]};
    if (k0 + 32 < DIM) {
      const int kn = k0 + 32;
#pragma unroll
      for (int mi = 0; mi < 2; ++mi)
        na[mi] = load_a_frag(Abase + (size_t)mi * 16 * DIM + kn, DIM, lane);
#pragma unroll
      for (int t = 0; t < 4; ++t)
        nb[t] = load_b_frag(Bbase + (size_t)t * 16 * DIM + kn, DIM, lane);
    }

#pragma unroll
    for (int t = 0; t < 4; ++t) {
      acc[0][t] = wmma_bf16(a[0], b[t], acc[0][t]);
      acc[1][t] = wmma_bf16(a[1], b[t], acc[1][t]);
    }
#pragma unroll
    for (int mi = 0; mi < 2; ++mi) a[mi] = na[mi];
#pragma unroll
    for (int t = 0; t < 4; ++t) b[t] = nb[t];
  }

#pragma unroll
  for (int t = 0; t < 4; ++t) {
    const int n = n0 + t * 16 + ln;
    const float pb = proj_b[n];
#pragma unroll
    for (int mi = 0; mi < 2; ++mi)
#pragma unroll
      for (int j = 0; j < 8; ++j) {
        const int row = m0 + mi * 16 + j + hf * 8;
        out[(size_t)row * DIM + n] = acc[mi][t][j] + pb;
      }
  }
}

// ---------------------------------------------------------------------------
extern "C" void kernel_launch(void* const* d_in, const int* in_sizes, int n_in,
                              void* d_out, int out_size, void* d_ws, size_t ws_size,
                              hipStream_t stream) {
  const float* x      = (const float*)d_in[0];
  const int*   mask   = (const int*)d_in[1];
  const float* qkv_w  = (const float*)d_in[2];
  const float* q_bias = (const float*)d_in[3];
  const float* v_bias = (const float*)d_in[4];
  const float* proj_w = (const float*)d_in[5];
  const float* proj_b = (const float*)d_in[6];
  float* out = (float*)d_out;

  // workspace layout (bytes), all regions 256B-aligned
  char* ws = (char*)d_ws;
  __bf16* xb = (__bf16*)(ws + 0);          // x bf16        : 12,582,912 B
  __bf16* wq = (__bf16*)(ws + 12582912);   // qkv_w bf16    :  3,538,944 B
  __bf16* wp = (__bf16*)(ws + 16121856);   // proj_w bf16   :  1,179,648 B
  __bf16* qs = (__bf16*)(ws + 17301504);   // q*scale [BH][N][hd]
  __bf16* kb = (__bf16*)(ws + 29884416);   // k       [BH][N][hd]
  __bf16* vt = (__bf16*)(ws + 42467328);   // v^T     [BH][hd][N]
  __bf16* ao = (__bf16*)(ws + 55050240);   // attn out [B*N][DIM]

  cvt_f32_bf16_kernel<<<1024, 256, 0, stream>>>(x, xb, BATCH * SEQ * DIM);
  cvt_f32_bf16_kernel<<<512, 256, 0, stream>>>(qkv_w, wq, 3 * DIM * DIM);
  cvt_f32_bf16_kernel<<<256, 256, 0, stream>>>(proj_w, wp, DIM * DIM);

  // QKV: M=8192 (y: 64 blocks of 128), N=2304 (x: 18 blocks of 128)
  qkv_gemm_kernel<<<dim3(18, 64), 256, 0, stream>>>(xb, wq, q_bias, v_bias,
                                                    qs, kb, vt);
  // Attention: B*H*(N/128) = 768 blocks, 8 waves each (16 q-rows per wave)
  attn_kernel<<<BATCH * NH * (SEQ / 128), 256, 0, stream>>>(qs, kb, vt, mask, ao);
  // Projection: N=768 (x: 6 blocks of 128), M=8192 (y: 64 blocks of 128)
  proj_gemm_kernel<<<dim3(6, 64), 256, 0, stream>>>(ao, wp, proj_b, out);
}
